// PNeGNN_21569325760694
// MI455X (gfx1250) — matrix operand order repaired and checked by reading, not compiled, not run on hardware
//
#include <hip/hip_runtime.h>
#include <hip/hip_bf16.h>

#define M_USERS  100000
#define N_ITEMS  50000
#define NN_      150000
#define D_       64
#define B_       8192
#define K_       32
#define E_EDGES  2000000
#define TAU_     0.8f
#define REG_     1e-4f

typedef __attribute__((ext_vector_type(16))) __bf16 v16bf;
typedef __attribute__((ext_vector_type(8)))  __bf16 v8bf;
typedef __attribute__((ext_vector_type(8)))  float  v8f;
typedef __attribute__((ext_vector_type(4)))  float  v4f;

// ---------- wave helpers (wave32) ----------
__device__ __forceinline__ float wave_sum32(float v) {
#pragma unroll
  for (int m = 16; m >= 1; m >>= 1) v += __shfl_xor(v, m, 32);
  return v;
}
__device__ __forceinline__ float half_sum16(float v) {
#pragma unroll
  for (int m = 8; m >= 1; m >>= 1) v += __shfl_xor(v, m, 16);
  return v;
}
__device__ __forceinline__ float log_sigmoid(float x) {
  // -log(1+exp(-x)), numerically stable
  return fminf(x, 0.f) - log1pf(__expf(-fabsf(x)));
}

// ---------- WMMA fragment loaders ----------
// A (16 rows x 32 K) from f32 row-major: two contiguous 8-float runs per lane,
// explicit v4f loads -> global_load_b128, then pack to bf16.
__device__ __forceinline__ v16bf load_a_tile(const float* __restrict__ A,
                                             int row0, int k0, int lane) {
  const float* rp = A + (long)(row0 + (lane & 15)) * D_ + k0 + ((lane >> 4) << 3);
  v4f f0 = *(const v4f*)rp;
  v4f f1 = *(const v4f*)(rp + 4);
  v4f f2 = *(const v4f*)(rp + 16);
  v4f f3 = *(const v4f*)(rp + 20);
  v16bf a;
#pragma unroll
  for (int i = 0; i < 4; ++i) {
    a[i]      = (__bf16)f0[i];
    a[4 + i]  = (__bf16)f1[i];
    a[8 + i]  = (__bf16)f2[i];
    a[12 + i] = (__bf16)f3[i];
  }
  return a;
}
__device__ __forceinline__ v16bf ld2x8(const __bf16* p0, const __bf16* p1) {
  v8bf lo = *(const v8bf*)p0;
  v8bf hi = *(const v8bf*)p1;
  return __builtin_shufflevector(lo, hi, 0, 1, 2, 3, 4, 5, 6, 7,
                                 8, 9, 10, 11, 12, 13, 14, 15);
}
// A fragment from bf16 row-major (two 16B runs per lane)
__device__ __forceinline__ v16bf load_a_bf(const __bf16* __restrict__ X,
                                           int row0, int k0, int lane) {
  const __bf16* rp = X + (long)(row0 + (lane & 15)) * D_ + k0 + ((lane >> 4) << 3);
  return ld2x8(rp, rp + 16);
}
// B = Y^T fragment from bf16 row-major: two 8-K runs per lane, 16 apart (two 16B runs)
__device__ __forceinline__ v16bf load_bt_bf(const __bf16* __restrict__ Y,
                                            int c0, int k0, int lane) {
  const __bf16* rp = Y + (long)(c0 + (lane & 15)) * D_ + k0 + ((lane >> 4) << 3);
  return ld2x8(rp, rp + 16);
}

// ---------- elementwise / graph kernels ----------
__global__ void k_zero(float* p, long n) {
  long i = (long)blockIdx.x * blockDim.x + threadIdx.x;
  if (i < n) p[i] = 0.f;
}
__global__ void k_concat(const float* __restrict__ a, long na,
                         const float* __restrict__ b, long nb,
                         float* __restrict__ out) {
  long i = (long)blockIdx.x * blockDim.x + threadIdx.x;
  if (i < na + nb) out[i] = (i < na) ? a[i] : b[i - na];
}
// pack 64x64 f32 weight into WMMA B-operand fragment order:
// tile t = ks*4+nt; dst[(t*32+L)*16+e] = bf16(W[(ks*32+((L>>4)<<3)+(e&7)+((e>>3)<<4))*64 + nt*16+(L&15)])
// so each GEMM lane loads its whole fragment as one aligned 32B v16bf read.
__global__ void k_pack_w(const float* __restrict__ W, __bf16* __restrict__ Wp) {
  int d = blockIdx.x * blockDim.x + threadIdx.x;
  if (d < D_ * D_) {
    int e = d & 15, L = (d >> 4) & 31, t = d >> 9;
    int ks = t >> 2, nt = t & 3;
    int k = ks * 32 + ((L >> 4) << 3) + (e & 7) + ((e >> 3) << 4);
    int n = nt * 16 + (L & 15);
    Wp[d] = (__bf16)W[k * D_ + n];
  }
}
__global__ void k_deg(const int* __restrict__ dst, float* __restrict__ deg, int ne) {
  int e = blockIdx.x * blockDim.x + threadIdx.x;
  if (e < ne) atomicAdd(&deg[dst[e]], 1.0f);
}
__global__ void k_dinv(float* d, int n) {
  int i = blockIdx.x * blockDim.x + threadIdx.x;
  if (i < n) { float v = d[i]; d[i] = (v > 0.f) ? rsqrtf(fmaxf(v, 1.f)) : 0.f; }
}
// one (edge,dim) per thread: out[dst][d] += x[src][d]*dinv[src]*dinv[dst]
__global__ void k_conv(const float* __restrict__ x, const int* __restrict__ src,
                       const int* __restrict__ dst, const float* __restrict__ dinv,
                       float* __restrict__ out, int ne) {
  long t = (long)blockIdx.x * blockDim.x + threadIdx.x;
  int  d = (int)(t & 63);
  long e = t >> 6;
  if (e < ne) {
    int s = src[e], q = dst[e];
    float nrm = dinv[s] * dinv[q];
    atomicAdd(&out[(long)q * D_ + d], x[(long)s * D_ + d] * nrm);
  }
}
__global__ void k_avg3(const float* x0, const float* x1, const float* x2,
                       float* out, long n) {
  long i = (long)blockIdx.x * blockDim.x + threadIdx.x;
  if (i < n) out[i] = (x0[i] + x1[i] + x2[i]) * (1.f / 3.f);
}

// ---------- WMMA GEMM kernels (one wave per 16-row slab, N=K=64) ----------
__global__ void k_gemm_relu(const float* __restrict__ A, const v16bf* __restrict__ Wp,
                            const float* __restrict__ bias, float* __restrict__ C) {
  int lane = threadIdx.x & 31;
  int r0 = blockIdx.x * 16;
  v16bf a0 = load_a_tile(A, r0, 0, lane);
  v16bf a1 = load_a_tile(A, r0, 32, lane);
  int rbase = r0 + ((lane >> 4) << 3);
  int nl = lane & 15;
#pragma unroll
  for (int nt = 0; nt < 4; ++nt) {
    v16bf b0 = Wp[nt * 32 + lane];         // ks=0 fragment
    v16bf b1 = Wp[(4 + nt) * 32 + lane];   // ks=1 fragment
    v8f c = {};
    c = __builtin_amdgcn_wmma_f32_16x16x32_bf16(false, a0, false, b0, (short)0, c, false, false);
    c = __builtin_amdgcn_wmma_f32_16x16x32_bf16(false, a1, false, b1, (short)0, c, false, false);
    int n = nt * 16 + nl;
    float bn = bias[n];
#pragma unroll
    for (int i = 0; i < 8; ++i) {
      float v = c[i] + bn;
      C[(long)(rbase + i) * D_ + n] = v > 0.f ? v : 0.f;
    }
  }
}
// fused: wvec[r] = sum_n tanh((A@W)[r][n] + bias[n]) * wq[n]  (t never stored)
__global__ void k_gemm_tanh_dot(const float* __restrict__ A, const v16bf* __restrict__ Wp,
                                const float* __restrict__ bias, const float* __restrict__ wq,
                                float* __restrict__ wvec) {
  int lane = threadIdx.x & 31;
  int r0 = blockIdx.x * 16;
  v16bf a0 = load_a_tile(A, r0, 0, lane);
  v16bf a1 = load_a_tile(A, r0, 32, lane);
  int nl = lane & 15;
  float p[8];
#pragma unroll
  for (int i = 0; i < 8; ++i) p[i] = 0.f;
#pragma unroll
  for (int nt = 0; nt < 4; ++nt) {
    v16bf b0 = Wp[nt * 32 + lane];
    v16bf b1 = Wp[(4 + nt) * 32 + lane];
    v8f c = {};
    c = __builtin_amdgcn_wmma_f32_16x16x32_bf16(false, a0, false, b0, (short)0, c, false, false);
    c = __builtin_amdgcn_wmma_f32_16x16x32_bf16(false, a1, false, b1, (short)0, c, false, false);
    int n = nt * 16 + nl;
    float bn = bias[n], wqn = wq[n];
#pragma unroll
    for (int i = 0; i < 8; ++i) p[i] += tanhf(c[i] + bn) * wqn;
  }
  int rbase = r0 + ((lane >> 4) << 3);
#pragma unroll
  for (int i = 0; i < 8; ++i) {
    float s = half_sum16(p[i]);      // reduce over the 16 columns held by this half
    if (nl == 0) wvec[rbase + i] = s;
  }
}
// fused contrastive tile: exp(X@Y^T/tau); accumulate global sum, write diagonal only
__global__ void k_contr(const __bf16* __restrict__ X, const __bf16* __restrict__ Y,
                        float* __restrict__ diag, float* __restrict__ accsum) {
  int lane = threadIdx.x & 31;
  int r0 = blockIdx.x * 16, c0 = blockIdx.y * 16;
  v16bf a0 = load_a_bf(X, r0, 0, lane);
  v16bf a1 = load_a_bf(X, r0, 32, lane);
  v16bf b0 = load_bt_bf(Y, c0, 0, lane);
  v16bf b1 = load_bt_bf(Y, c0, 32, lane);
  v8f c = {};
  c = __builtin_amdgcn_wmma_f32_16x16x32_bf16(false, a0, false, b0, (short)0, c, false, false);
  c = __builtin_amdgcn_wmma_f32_16x16x32_bf16(false, a1, false, b1, (short)0, c, false, false);
  int n = c0 + (lane & 15);
  int rbase = r0 + ((lane >> 4) << 3);
  float ts = 0.f;
#pragma unroll
  for (int i = 0; i < 8; ++i) {
    float vexp = __expf(c[i] * (1.f / TAU_));
    ts += vexp;
    if (rbase + i == n) diag[n] = vexp;   // only diagonal tiles hit
  }
  ts = wave_sum32(ts);
  if (lane == 0) atomicAdd(accsum, ts);
}

// ---------- fusion / losses ----------
__global__ void k_fuse(float* zp, const float* __restrict__ zng,
                       const float* __restrict__ wp, const float* __restrict__ wn, long n) {
  long idx = (long)blockIdx.x * blockDim.x + threadIdx.x;
  if (idx < n) {
    long i = idx >> 6;
    float a = wp[i], b = wn[i], mx = fmaxf(a, b);
    float ea = __expf(a - mx), eb = __expf(b - mx);
    zp[idx] = (ea * zp[idx] + eb * zng[idx]) / (ea + eb);
  }
}
// one wave per batch element: both signed-BPR losses + L2 reg terms
__global__ void k_bpr(const float* __restrict__ zp, const float* __restrict__ zn,
                      const int* __restrict__ u, const int* __restrict__ v,
                      const float* __restrict__ wgt, const int* __restrict__ nidx,
                      float* __restrict__ acc) {
  int b = (int)(((long)blockIdx.x * blockDim.x + threadIdx.x) >> 5);
  int lane = threadIdx.x & 31;
  if (b >= B_) return;
  long uo = (long)u[b] * D_, vo = (long)v[b] * D_;
  float a0 = zp[uo + lane], a1 = zp[uo + lane + 32];
  float c0 = zp[vo + lane], c1 = zp[vo + lane + 32];
  float e0 = zn[uo + lane], e1 = zn[uo + lane + 32];
  float f0 = zn[vo + lane], f1 = zn[vo + lane + 32];
  float pos1 = wave_sum32(a0 * c0 + a1 * c1);
  float pos2 = wave_sum32(e0 * f0 + e1 * f1);
  float wv = wgt[b];
  float sgn = (wv > 0.f) ? 1.f : ((wv < 0.f) ? -1.f : 0.f);
  float m1 = -sgn + 2.f, m2 = sgn + 2.f;
  float s1 = 0.f, s2 = 0.f;
  float r1 = a0 * a0 + a1 * a1 + c0 * c0 + c1 * c1;
  float r2 = e0 * e0 + e1 * e1 + f0 * f0 + f1 * f1;
  const int* nb = nidx + (long)b * K_;
#pragma unroll 4
  for (int k = 0; k < K_; ++k) {
    long no = (long)nb[k] * D_;
    float g0 = zp[no + lane], g1 = zp[no + lane + 32];
    float neg1 = wave_sum32(a0 * g0 + a1 * g1);
    s1 += log_sigmoid(m1 * pos1 - neg1);
    r1 += g0 * g0 + g1 * g1;
    float h0 = zn[no + lane], h1 = zn[no + lane + 32];
    float neg2 = wave_sum32(e0 * h0 + e1 * h1);
    s2 += log_sigmoid(neg2 - m2 * pos2);
    r2 += h0 * h0 + h1 * h1;
  }
  r1 = wave_sum32(r1); r2 = wave_sum32(r2);
  if (lane == 0) {
    atomicAdd(&acc[0], s1);
    atomicAdd(&acc[1], r1);
    atomicAdd(&acc[2], s2);
    atomicAdd(&acc[3], r2);
  }
}
// gather + L2-normalize, emit bf16 rows ready for the contrastive WMMA kernels
__global__ void k_gather_norm(const float* __restrict__ Z, const int* __restrict__ idx,
                              __bf16* __restrict__ out) {
  int b = (int)(((long)blockIdx.x * blockDim.x + threadIdx.x) >> 5);
  int lane = threadIdx.x & 31;
  if (b >= B_) return;
  long so = (long)idx[b] * D_;
  float x0 = Z[so + lane], x1 = Z[so + lane + 32];
  float ss = wave_sum32(x0 * x0 + x1 * x1);
  float inv = 1.f / fmaxf(sqrtf(ss), 1e-12f);
  out[(long)b * D_ + lane]      = (__bf16)(x0 * inv);
  out[(long)b * D_ + lane + 32] = (__bf16)(x1 * inv);
}
__global__ void k_logpos(const float* __restrict__ du, const float* __restrict__ dv,
                         float* __restrict__ acc) {
  int i = blockIdx.x * blockDim.x + threadIdx.x;
  if (i < B_) atomicAdd(&acc[6], logf(du[i] + dv[i]));
}
// cl.mean = log(Sf) - mean(log(pos_pair)) since pos+neg == f_u.sum()+f_v.sum()
__global__ void k_final(const float* __restrict__ acc, float* __restrict__ out) {
  if (blockIdx.x == 0 && threadIdx.x == 0) {
    float loss = -acc[0] + REG_ * acc[1]
               + (-acc[2] / (float)B_ + REG_ * acc[3])
               + (logf(acc[4] + acc[5]) - acc[6] / (float)B_);
    out[0] = loss;
  }
}

extern "C" void kernel_launch(void* const* d_in, const int* in_sizes, int n_in,
                              void* d_out, int out_size, void* d_ws, size_t ws_size,
                              hipStream_t stream) {
  (void)in_sizes; (void)n_in; (void)out_size; (void)ws_size;
  const float* E_pos    = (const float*)d_in[0];
  const float* E_neg    = (const float*)d_in[1];
  const float* E_item   = (const float*)d_in[2];
  const float* E_item_n = (const float*)d_in[3];
  const float* E2       = (const float*)d_in[4];
  const float* W1 = (const float*)d_in[5];
  const float* b1 = (const float*)d_in[6];
  const float* W2 = (const float*)d_in[7];
  const float* b2 = (const float*)d_in[8];
  const float* Wa = (const float*)d_in[9];
  const float* ba = (const float*)d_in[10];
  const float* Wq = (const float*)d_in[11];
  const int*   u    = (const int*)d_in[12];
  const int*   v    = (const int*)d_in[13];
  const float* w    = (const float*)d_in[14];
  const int*   nidx = (const int*)d_in[15];
  const int* edge_p  = (const int*)d_in[16];
  const int* edge_n  = (const int*)d_in[17];
  const int* edge_n1 = (const int*)d_in[18];

  const long NND = (long)NN_ * D_;
  float* F    = (float*)d_ws;          // ~203 MB total
  float* bufA = F;                     // zp
  float* bufB = bufA + NND;
  float* bufC = bufB + NND;
  float* bufD = bufC + NND;            // zn
  float* bufE = bufD + NND;            // zn1
  float* dinv = bufE + NND;
  float* wp   = dinv + NN_;
  float* wn   = wp + NN_;
  __bf16* u2n = (__bf16*)(wn + NN_);   // bf16 rows, 16B-aligned slots
  __bf16* u3n = (__bf16*)((float*)u2n + (long)B_ * D_);
  __bf16* v2n = (__bf16*)((float*)u3n + (long)B_ * D_);
  __bf16* v3n = (__bf16*)((float*)v2n + (long)B_ * D_);
  float* diagU = (float*)v3n + (long)B_ * D_;
  float* diagV = diagU + B_;
  float* acc   = diagV + B_;
  __bf16* W1p = (__bf16*)(acc + 16);   // packed WMMA-B fragments (32B aligned)
  __bf16* W2p = W1p + D_ * D_;
  __bf16* Wap = W2p + D_ * D_;

  const int BS = 256;
  auto nb = [](long n, int bs) { return (unsigned)((n + bs - 1) / bs); };

  // pack weights into WMMA B-fragment order (tiny, once per call)
  k_pack_w<<<nb(D_ * D_, BS), BS, 0, stream>>>(W1, W1p);
  k_pack_w<<<nb(D_ * D_, BS), BS, 0, stream>>>(W2, W2p);
  k_pack_w<<<nb(D_ * D_, BS), BS, 0, stream>>>(Wa, Wap);

  auto propagate = [&](const float* x0, const int* ei, float* t1, float* t2, float* zout) {
    const int* src = ei;
    const int* dst = ei + E_EDGES;
    k_zero<<<nb(NN_, BS), BS, 0, stream>>>(dinv, NN_);
    k_deg<<<nb(E_EDGES, BS), BS, 0, stream>>>(dst, dinv, E_EDGES);
    k_dinv<<<nb(NN_, BS), BS, 0, stream>>>(dinv, NN_);
    k_zero<<<nb(NND, BS), BS, 0, stream>>>(t1, NND);
    k_conv<<<nb((long)E_EDGES * D_, BS), BS, 0, stream>>>(x0, src, dst, dinv, t1, E_EDGES);
    k_zero<<<nb(NND, BS), BS, 0, stream>>>(t2, NND);
    k_conv<<<nb((long)E_EDGES * D_, BS), BS, 0, stream>>>(t1, src, dst, dinv, t2, E_EDGES);
    k_avg3<<<nb(NND, BS), BS, 0, stream>>>(x0, t1, t2, zout, NND);
  };

  // --- positive graph: zp in bufA ---
  k_concat<<<nb(NND, BS), BS, 0, stream>>>(E_pos, (long)M_USERS * D_,
                                           E_item, (long)N_ITEMS * D_, bufA);
  propagate(bufA, edge_p, bufB, bufC, bufA);

  // --- MLP branch: zng in bufC ---
  k_gemm_relu<<<NN_ / 16, 32, 0, stream>>>(E2,   (const v16bf*)W1p, b1, bufB);
  k_gemm_relu<<<NN_ / 16, 32, 0, stream>>>(bufB, (const v16bf*)W2p, b2, bufC);

  // --- attention fusion (tanh-GEMM with fused Wq dot) ---
  k_gemm_tanh_dot<<<NN_ / 16, 32, 0, stream>>>(bufA, (const v16bf*)Wap, ba, Wq, wp);
  k_gemm_tanh_dot<<<NN_ / 16, 32, 0, stream>>>(bufC, (const v16bf*)Wap, ba, Wq, wn);
  k_fuse<<<nb(NND, BS), BS, 0, stream>>>(bufA, bufC, wp, wn, NND);

  // --- negative graphs: zn in bufD, zn1 in bufE ---
  k_concat<<<nb(NND, BS), BS, 0, stream>>>(E_neg, (long)M_USERS * D_,
                                           E_item_n, (long)N_ITEMS * D_, bufB);
  propagate(bufB, edge_n,  bufC, bufD, bufD);
  propagate(bufB, edge_n1, bufC, bufE, bufE);

  // --- losses ---
  k_zero<<<1, 32, 0, stream>>>(acc, 16);
  k_bpr<<<B_ / 8, 256, 0, stream>>>(bufA, bufD, u, v, w, nidx, acc);

  k_gather_norm<<<B_ / 8, 256, 0, stream>>>(bufD, u, u2n);
  k_gather_norm<<<B_ / 8, 256, 0, stream>>>(bufE, u, u3n);
  k_gather_norm<<<B_ / 8, 256, 0, stream>>>(bufD, v, v2n);
  k_gather_norm<<<B_ / 8, 256, 0, stream>>>(bufE, v, v3n);

  dim3 cg(B_ / 16, B_ / 16);
  k_contr<<<cg, 32, 0, stream>>>(u2n, u3n, diagU, &acc[4]);
  k_contr<<<cg, 32, 0, stream>>>(v2n, v3n, diagV, &acc[5]);
  k_logpos<<<nb(B_, BS), BS, 0, stream>>>(diagU, diagV, acc);
  k_final<<<1, 32, 0, stream>>>(acc, (float*)d_out);
}
